// ModernTransformer_48438641164769
// MI455X (gfx1250) — matrix operand order repaired and checked by reading, not compiled, not run on hardware
//
#include <hip/hip_runtime.h>
#include <stdint.h>

#define USE_ASYNC_LDS 1

// ---------------------------------------------------------------------------
// Types / fragments
// ---------------------------------------------------------------------------
typedef __bf16 bf16_t;
typedef bf16_t v16bf __attribute__((ext_vector_type(16)));
typedef float  v8f   __attribute__((ext_vector_type(8)));
typedef unsigned int u32x4 __attribute__((ext_vector_type(4)));

union Frag16 {            // 16 bf16 elements = 8 VGPRs (A or B fragment)
    v16bf v;
    u32x4 q[2];
    unsigned short h[16];
};

// Model shape constants (match the reference)
constexpr int NB  = 2;        // batch
constexpr int NT  = 1024;     // seq len
constexpr int ND  = 1024;     // model dim
constexpr int NH  = 16;       // query heads
constexpr int NKVH= 4;        // kv heads
constexpr int NHD = 64;       // head dim
constexpr int NKV = NKVH*NHD; // 256
constexpr int NF  = 4096;     // ffn hidden
constexpr int NVOC= 32000;
constexpr int NL  = 4;
constexpr int NBT = NB*NT;    // 2048 rows

#define DEV __device__ __forceinline__

DEV unsigned short f32_to_bf16(float f) {
    union { float f; uint32_t u; } x; x.f = f;
    uint32_t r = x.u + 0x7FFFu + ((x.u >> 16) & 1u);   // round-to-nearest-even
    return (unsigned short)(r >> 16);
}
DEV float bf16_to_f32(unsigned short h) {
    union { uint32_t u; float f; } x; x.u = ((uint32_t)h) << 16;
    return x.f;
}

DEV v8f wmma_bf16(const Frag16& a, const Frag16& b, v8f c) {
    return __builtin_amdgcn_wmma_f32_16x16x32_bf16(
        false, a.v, false, b.v, (short)0, c, false, false);
}

// LDS byte offset of a __shared__ object: generic pointers to LDS carry the
// LDS byte address in their low 32 bits on CDNA5 (aperture in the high bits).
DEV unsigned lds_addr(const void* p) { return (unsigned)(size_t)p; }

// ---------------------------------------------------------------------------
// Elementwise kernels
// ---------------------------------------------------------------------------
__global__ __launch_bounds__(256) void to_bf16_kernel(
    const float* __restrict__ in, unsigned short* __restrict__ out, size_t n)
{
    size_t i = (size_t)blockIdx.x * 256 + threadIdx.x;
    if (i < n) out[i] = f32_to_bf16(in[i]);
}

__global__ __launch_bounds__(256) void embed_kernel(
    const int* __restrict__ ids, const float* __restrict__ emb,
    float* __restrict__ x)
{
    size_t i = (size_t)blockIdx.x * 256 + threadIdx.x;
    if (i >= (size_t)NBT * ND) return;
    size_t row = i / ND;  int d = (int)(i % ND);
    x[i] = emb[(size_t)ids[row] * ND + d];
}

__global__ __launch_bounds__(256) void rmsnorm_kernel(
    const float* __restrict__ x, const float* __restrict__ w,
    unsigned short* __restrict__ out)
{
    int row = blockIdx.x;
    const float* xr = x + (size_t)row * ND;
    float ss = 0.f;
    for (int i = threadIdx.x; i < ND; i += 256) { float v = xr[i]; ss += v * v; }
    __shared__ float red[256];
    red[threadIdx.x] = ss; __syncthreads();
    for (int s = 128; s > 0; s >>= 1) {
        if (threadIdx.x < s) red[threadIdx.x] += red[threadIdx.x + s];
        __syncthreads();
    }
    float inv = rsqrtf(red[0] / (float)ND + 1e-6f);
    for (int i = threadIdx.x; i < ND; i += 256)
        out[(size_t)row * ND + i] = f32_to_bf16(xr[i] * inv * w[i]);
}

// RoPE in-place on bf16 activations laid out [row][head*64 + d]
__global__ __launch_bounds__(256) void rope_kernel(
    unsigned short* __restrict__ x, int nheads)
{
    size_t total = (size_t)NBT * nheads * (NHD / 2);
    size_t i = (size_t)blockIdx.x * 256 + threadIdx.x;
    if (i >= total) return;
    int d = (int)(i % (NHD / 2));
    size_t tmp = i / (NHD / 2);
    int hh = (int)(tmp % nheads);
    size_t row = tmp / nheads;
    int t = (int)(row % NT);
    float invf = __powf(10000.f, -((float)(2 * d)) / (float)NHD);
    float ang = (float)t * invf;
    float s, c; __sincosf(ang, &s, &c);
    size_t base = row * (size_t)(nheads * NHD) + (size_t)hh * NHD;
    float x1 = bf16_to_f32(x[base + d]);
    float x2 = bf16_to_f32(x[base + d + NHD / 2]);
    x[base + d]            = f32_to_bf16(x1 * c - x2 * s);
    x[base + d + NHD / 2]  = f32_to_bf16(x2 * c + x1 * s);
}

// ---------------------------------------------------------------------------
// Generic bf16 WMMA GEMM:  C[M,N] = A[M,K] * W[N,K]^T  (+ epilogue)
// One wave computes a 32x64 strip (2 M-tiles x 4 N-tiles): each B fragment is
// reused by two WMMAs -> 1.5 b128 loads per WMMA. M is always NBT.
// EPI: 0=store f32, 1=store bf16, 2=resid-add f32, 3=silu(aux)*c -> bf16
// ---------------------------------------------------------------------------
template<int EPI>
__global__ __launch_bounds__(128) void gemm_kernel(
    const unsigned short* __restrict__ A, int lda,
    const unsigned short* __restrict__ W, int ldw,
    const float* aux, float* outf, unsigned short* outh,
    int N, int K)
{
    const int lane = threadIdx.x & 31;
    const int wave = threadIdx.x >> 5;
    const int r16  = lane & 15;
    const int half = lane >> 4;
    const int m0   = (blockIdx.y * 4 + wave) * 32;
    const long n0  = (long)blockIdx.x * 64;

    v8f acc[2][4] = {};
    // A fragment: lane holds row, K-chunks {h*8..h*8+7, h*8+16..h*8+23}
    const unsigned short* ar0 = A + (size_t)(m0 + r16) * lda + half * 8;
    const unsigned short* ar1 = ar0 + (size_t)16 * lda;
    // B fragment: lane holds column n, K-chunks {h*8..h*8+7, h*8+16..h*8+23}
    const unsigned short* wr[4];
#pragma unroll
    for (int t = 0; t < 4; ++t)
        wr[t] = W + (size_t)(n0 + t * 16 + r16) * ldw + half * 8;

    for (int k0 = 0; k0 < K; k0 += 32) {
        Frag16 a0, a1;
        a0.q[0] = *(const u32x4*)(ar0 + k0);
        a0.q[1] = *(const u32x4*)(ar0 + k0 + 16);
        a1.q[0] = *(const u32x4*)(ar1 + k0);
        a1.q[1] = *(const u32x4*)(ar1 + k0 + 16);
#pragma unroll
        for (int t = 0; t < 4; ++t) {
            Frag16 b;
            b.q[0] = *(const u32x4*)(wr[t] + k0);
            b.q[1] = *(const u32x4*)(wr[t] + k0 + 16);
            acc[0][t] = wmma_bf16(a0, b, acc[0][t]);
            acc[1][t] = wmma_bf16(a1, b, acc[1][t]);
        }
    }

#pragma unroll
    for (int mi = 0; mi < 2; ++mi) {
        const int mbase = m0 + mi * 16 + half * 8;
#pragma unroll
        for (int t = 0; t < 4; ++t) {
            long col = n0 + t * 16 + r16;
#pragma unroll
            for (int j = 0; j < 8; ++j) {
                size_t idx = (size_t)(mbase + j) * (size_t)N + (size_t)col;
                float c = acc[mi][t][j];
                if (EPI == 0) {
                    outf[idx] = c;
                } else if (EPI == 1) {
                    outh[idx] = f32_to_bf16(c);
                } else if (EPI == 2) {
                    outf[idx] = aux[idx] + c;
                } else { // 3: aux holds raw gate f32; apply silu and multiply
                    float g = aux[idx];
                    float sg = g / (1.f + __expf(-g));
                    outh[idx] = f32_to_bf16(sg * c);
                }
            }
        }
    }
}

// ---------------------------------------------------------------------------
// Flash-style causal attention. 4 waves/block cover 64 consecutive queries of
// one (b,h); each 32-key x 64-d K/V tile is staged into LDS ONCE per block via
// async global->LDS DMA (ASYNCcnt) and consumed by all 4 waves.
// q: [row][h*64+d] bf16 (stride ND); k,v: [row][hkv*64+d] (stride NKV)
// ---------------------------------------------------------------------------
__global__ __launch_bounds__(128) void attn_kernel(
    const unsigned short* __restrict__ q,
    const unsigned short* __restrict__ k,
    const unsigned short* __restrict__ v,
    unsigned short* __restrict__ o)
{
    const int qb64 = blockIdx.x;        // 0..15 : 64-query super-tile
    const int h  = blockIdx.y;          // 0..15
    const int b  = blockIdx.z;          // 0..1
    const int tid  = threadIdx.x;
    const int wave = tid >> 5;
    const int lane = tid & 31;
    const int r16  = lane & 15;
    const int half = lane >> 4;
    const int q0   = qb64 * 64 + wave * 16;   // this wave's query tile
    const int hkv  = h / (NH / NKVH);

    __shared__ __align__(16) unsigned short Klds[32 * 64];   // [key][d]
    __shared__ __align__(16) unsigned short Vlds[32 * 64];   // [key][d]
    __shared__ __align__(16) unsigned short pshm[4][16 * 32];

    const unsigned short* qg = q + (size_t)(b * NT) * ND  + (size_t)h   * NHD;
    const unsigned short* kg = k + (size_t)(b * NT) * NKV + (size_t)hkv * NHD;
    const unsigned short* vg = v + (size_t)(b * NT) * NKV + (size_t)hkv * NHD;

    // Q fragments for K-dim slices [0,32) and [32,64)
    Frag16 qa[2];
    const unsigned short* qrow = qg + (size_t)(q0 + r16) * ND;
#pragma unroll
    for (int s = 0; s < 2; ++s) {
        int kbs = s * 32 + half * 8;
        qa[s].q[0] = *(const u32x4*)(qrow + kbs);
        qa[s].q[1] = *(const u32x4*)(qrow + kbs + 16);
    }

    float mstat[8], lstat[8];
#pragma unroll
    for (int j = 0; j < 8; ++j) { mstat[j] = -1e30f; lstat[j] = 0.f; }
    v8f oacc[4] = {};

    const float scale = 0.125f;               // 1/sqrt(64)
    const int nkb = (qb64 * 64 + 63) / 32 + 1;

    for (int kv0 = 0; kv0 < nkb * 32; kv0 += 32) {
        // ---- cooperative K/V tile stage: 256 16B chunks each, 128 threads ----
#pragma unroll
        for (int c = tid; c < 256; c += 128) {
            int key = c >> 3, d8 = (c & 7) * 8;
            const unsigned short* kp = kg + (size_t)(kv0 + key) * NKV + d8;
            const unsigned short* vp = vg + (size_t)(kv0 + key) * NKV + d8;
#if USE_ASYNC_LDS
            unsigned koff = lds_addr(&Klds[key * 64 + d8]);
            unsigned voff = lds_addr(&Vlds[key * 64 + d8]);
            asm volatile("global_load_async_to_lds_b128 %0, %1, off"
                         :: "v"(koff), "v"((unsigned long long)(size_t)kp)
                         : "memory");
            asm volatile("global_load_async_to_lds_b128 %0, %1, off"
                         :: "v"(voff), "v"((unsigned long long)(size_t)vp)
                         : "memory");
#else
            *(u32x4*)&Klds[key * 64 + d8] = *(const u32x4*)kp;
            *(u32x4*)&Vlds[key * 64 + d8] = *(const u32x4*)vp;
#endif
        }
#if USE_ASYNC_LDS
        asm volatile("s_wait_asynccnt 0x0" ::: "memory");
#endif
        __syncthreads();

        const bool active = (kv0 <= q0 + 15);   // uniform per wave
        if (active) {
            // ---- scores: S(16x32) = Q(16x64) x K^T  (K B-frags from LDS) ----
            v8f s[2] = {};
#pragma unroll
            for (int t = 0; t < 2; ++t) {
#pragma unroll
                for (int ss = 0; ss < 2; ++ss) {
                    Frag16 bk;
                    const unsigned short* kr =
                        &Klds[(t * 16 + r16) * 64 + ss * 32 + half * 8];
                    bk.q[0] = *(const u32x4*)(kr);
                    bk.q[1] = *(const u32x4*)(kr + 16);
                    s[t] = wmma_bf16(qa[ss], bk, s[t]);
                }
            }
            // ---- mask + scale ----
#pragma unroll
            for (int t = 0; t < 2; ++t) {
#pragma unroll
                for (int j = 0; j < 8; ++j) {
                    int key = kv0 + t * 16 + r16;
                    int qrowg = q0 + j + 8 * half;
                    float val = s[t][j] * scale;
                    s[t][j] = (key > qrowg) ? -1e30f : val;
                }
            }
            // ---- online softmax (one row = one half-wave, one acc slot) ----
            float corr[8];
#pragma unroll
            for (int j = 0; j < 8; ++j) {
                float mx = fmaxf(s[0][j], s[1][j]);
#pragma unroll
                for (int off = 1; off < 16; off <<= 1)
                    mx = fmaxf(mx, __shfl_xor(mx, off, 32));
                float nm = fmaxf(mstat[j], mx);
                corr[j] = __expf(mstat[j] - nm);
                float p0 = __expf(s[0][j] - nm);
                float p1 = __expf(s[1][j] - nm);
                s[0][j] = p0; s[1][j] = p1;
                float rs = p0 + p1;
#pragma unroll
                for (int off = 1; off < 16; off <<= 1)
                    rs += __shfl_xor(rs, off, 32);
                lstat[j] = lstat[j] * corr[j] + rs;
                mstat[j] = nm;
            }
            // ---- P (C-layout) -> LDS row-major 16x32 bf16, wave-private ----
            unsigned short* ps = pshm[wave];
#pragma unroll
            for (int t = 0; t < 2; ++t)
#pragma unroll
                for (int j = 0; j < 8; ++j)
                    ps[(j + 8 * half) * 32 + (r16 + 16 * t)] = f32_to_bf16(s[t][j]);
            // wave-private LDS RAW: drain DScnt (can't __syncthreads here,
            // this branch is wave-divergent across the block)
            asm volatile("s_wait_dscnt 0x0" ::: "memory");
            Frag16 pa;
            pa.q[0] = *(const u32x4*)(ps + r16 * 32 + half * 16);
            pa.q[1] = *(const u32x4*)(ps + r16 * 32 + half * 16 + 8);
            // ---- O = O*corr + P x V  (V B-frags from LDS) ----
#pragma unroll
            for (int t = 0; t < 4; ++t) {
#pragma unroll
                for (int j = 0; j < 8; ++j) oacc[t][j] *= corr[j];
                Frag16 bv;   // B[kk][n] = Vlds[half*16 + kk][t*16 + r16]
                const unsigned short* vcol = &Vlds[(half * 16) * 64 + t * 16 + r16];
#pragma unroll
                for (int j = 0; j < 8; ++j) {
                    bv.h[2 * j]     = vcol[(2 * j) * 64];
                    bv.h[2 * j + 1] = vcol[(2 * j + 1) * 64];
                }
                oacc[t] = wmma_bf16(pa, bv, oacc[t]);
            }
        }
        __syncthreads();   // protect Klds/Vlds reuse next iteration
    }

    // ---- normalize + store bf16 to [row][h*64+d] ----
#pragma unroll
    for (int t = 0; t < 4; ++t) {
#pragma unroll
        for (int j = 0; j < 8; ++j) {
            float val = oacc[t][j] / lstat[j];
            size_t row = (size_t)(b * NT + q0 + j + 8 * half);
            size_t col = (size_t)h * NHD + t * 16 + r16;
            o[row * ND + col] = f32_to_bf16(val);
        }
    }
}

// ---------------------------------------------------------------------------
// Host orchestration
// ---------------------------------------------------------------------------
static void launch_gemm(int epi,
                        const unsigned short* A, int lda,
                        const unsigned short* W, int ldw,
                        const float* aux, float* outf, unsigned short* outh,
                        int N, int K, hipStream_t stream)
{
    dim3 grid(N / 64, NBT / 128);   // 32 rows/wave * 4 waves = 128 rows/block
    dim3 block(128);
    switch (epi) {
    case 0: gemm_kernel<0><<<grid, block, 0, stream>>>(A, lda, W, ldw, aux, outf, outh, N, K); break;
    case 1: gemm_kernel<1><<<grid, block, 0, stream>>>(A, lda, W, ldw, aux, outf, outh, N, K); break;
    case 2: gemm_kernel<2><<<grid, block, 0, stream>>>(A, lda, W, ldw, aux, outf, outh, N, K); break;
    default:gemm_kernel<3><<<grid, block, 0, stream>>>(A, lda, W, ldw, aux, outf, outh, N, K); break;
    }
}

static void launch_to_bf16(const float* in, unsigned short* out, size_t n, hipStream_t s)
{
    to_bf16_kernel<<<(unsigned)((n + 255) / 256), 256, 0, s>>>(in, out, n);
}

extern "C" void kernel_launch(void* const* d_in, const int* in_sizes, int n_in,
                              void* d_out, int out_size, void* d_ws, size_t ws_size,
                              hipStream_t stream)
{
    const int*   ids  = (const int*)  d_in[0];
    const float* embw = (const float*)d_in[1];
    const float* ln1  = (const float*)d_in[2];
    const float* qw   = (const float*)d_in[3];
    const float* kw   = (const float*)d_in[4];
    const float* vw   = (const float*)d_in[5];
    const float* ow   = (const float*)d_in[6];
    const float* ln2  = (const float*)d_in[7];
    const float* w1   = (const float*)d_in[8];
    const float* w2   = (const float*)d_in[9];
    const float* w3   = (const float*)d_in[10];
    const float* lnf  = (const float*)d_in[11];

    // ---- workspace carve ----
    uint8_t* p = (uint8_t*)d_ws;
    auto allocH = [&](size_t n) { unsigned short* r = (unsigned short*)p;
                                  p += ((n * 2 + 255) / 256) * 256; return r; };
    auto allocF = [&](size_t n) { float* r = (float*)p;
                                  p += ((n * 4 + 255) / 256) * 256; return r; };

    unsigned short* qw_h  = allocH((size_t)NL * ND * ND);
    unsigned short* kw_h  = allocH((size_t)NL * NKV * ND);
    unsigned short* vw_h  = allocH((size_t)NL * NKV * ND);
    unsigned short* ow_h  = allocH((size_t)NL * ND * ND);
    unsigned short* w1_h  = allocH((size_t)NL * NF * ND);
    unsigned short* w2_h  = allocH((size_t)NL * ND * NF);
    unsigned short* w3_h  = allocH((size_t)NL * NF * ND);
    unsigned short* emb_h = allocH((size_t)NVOC * ND);
    float*          x     = allocF((size_t)NBT * ND);
    unsigned short* hbuf  = allocH((size_t)NBT * ND);
    unsigned short* qbuf  = allocH((size_t)NBT * ND);
    unsigned short* kbuf  = allocH((size_t)NBT * NKV);
    unsigned short* vbuf  = allocH((size_t)NBT * NKV);
    unsigned short* abuf  = allocH((size_t)NBT * ND);
    float*          gate  = allocF((size_t)NBT * NF);
    unsigned short* gu    = allocH((size_t)NBT * NF);

    // ---- weight conversions (bf16 is the WMMA-fed precision) ----
    launch_to_bf16(qw,   qw_h,  (size_t)NL * ND * ND,  stream);
    launch_to_bf16(kw,   kw_h,  (size_t)NL * NKV * ND, stream);
    launch_to_bf16(vw,   vw_h,  (size_t)NL * NKV * ND, stream);
    launch_to_bf16(ow,   ow_h,  (size_t)NL * ND * ND,  stream);
    launch_to_bf16(w1,   w1_h,  (size_t)NL * NF * ND,  stream);
    launch_to_bf16(w2,   w2_h,  (size_t)NL * ND * NF,  stream);
    launch_to_bf16(w3,   w3_h,  (size_t)NL * NF * ND,  stream);
    launch_to_bf16(embw, emb_h, (size_t)NVOC * ND,     stream);

    // ---- embedding gather (fp32 residual stream) ----
    embed_kernel<<<(unsigned)(((size_t)NBT * ND + 255) / 256), 256, 0, stream>>>(ids, embw, x);

    for (int i = 0; i < NL; ++i) {
        const unsigned short* wqi = qw_h + (size_t)i * ND * ND;
        const unsigned short* wki = kw_h + (size_t)i * NKV * ND;
        const unsigned short* wvi = vw_h + (size_t)i * NKV * ND;
        const unsigned short* woi = ow_h + (size_t)i * ND * ND;
        const unsigned short* w1i = w1_h + (size_t)i * NF * ND;
        const unsigned short* w2i = w2_h + (size_t)i * ND * NF;
        const unsigned short* w3i = w3_h + (size_t)i * NF * ND;

        // attention block
        rmsnorm_kernel<<<NBT, 256, 0, stream>>>(x, ln1 + (size_t)i * ND, hbuf);
        launch_gemm(1, hbuf, ND, wqi, ND, nullptr, nullptr, qbuf, ND,  ND, stream);
        launch_gemm(1, hbuf, ND, wki, ND, nullptr, nullptr, kbuf, NKV, ND, stream);
        launch_gemm(1, hbuf, ND, wvi, ND, nullptr, nullptr, vbuf, NKV, ND, stream);
        {
            size_t nq = (size_t)NBT * NH   * (NHD / 2);
            size_t nk = (size_t)NBT * NKVH * (NHD / 2);
            rope_kernel<<<(unsigned)((nq + 255) / 256), 256, 0, stream>>>(qbuf, NH);
            rope_kernel<<<(unsigned)((nk + 255) / 256), 256, 0, stream>>>(kbuf, NKVH);
        }
        attn_kernel<<<dim3(NT / 64, NH, NB), 128, 0, stream>>>(qbuf, kbuf, vbuf, abuf);
        launch_gemm(2, abuf, ND, woi, ND, x, x, nullptr, ND, ND, stream);

        // mlp block
        rmsnorm_kernel<<<NBT, 256, 0, stream>>>(x, ln2 + (size_t)i * ND, hbuf);
        launch_gemm(0, hbuf, ND, w1i, ND, nullptr, gate, nullptr, NF, ND, stream); // raw gate f32
        launch_gemm(3, hbuf, ND, w3i, ND, gate, nullptr, gu, NF, ND, stream);      // silu(g)*up -> bf16
        launch_gemm(2, gu, NF, w2i, NF, x, x, nullptr, ND, NF, stream);            // + residual
    }

    // final norm + logits
    rmsnorm_kernel<<<NBT, 256, 0, stream>>>(x, lnf, hbuf);
    launch_gemm(0, hbuf, ND, emb_h, ND, nullptr, (float*)d_out, nullptr, NVOC, ND, stream);
}
